// TopoFlowAttentionV2_83665962926649
// MI455X (gfx1250) — matrix operand order repaired and checked by reading, not compiled, not run on hardware
//
#include <hip/hip_runtime.h>
#include <hip/hip_bf16.h>

typedef __attribute__((ext_vector_type(16))) __bf16 v16bf;
typedef __attribute__((ext_vector_type(8)))  __bf16 v8bf;
typedef __attribute__((ext_vector_type(4)))  __bf16 v4bf;
typedef __attribute__((ext_vector_type(8)))  float  v8f;

#define B_SZ     4
#define N_SEQ    1024
#define D_MODEL  768
#define N_HEADS  12
#define HEAD_DIM 64
#define NBUCK    32

// ---------------------------------------------------------------------------
// helpers
// ---------------------------------------------------------------------------
static __device__ __forceinline__ v16bf pack16(const __bf16* lo, const __bf16* hi) {
  v8bf a = *(const v8bf*)lo;
  v8bf b = *(const v8bf*)hi;
  v16bf r;
#pragma unroll
  for (int i = 0; i < 8; ++i) { r[i] = a[i]; r[i + 8] = b[i]; }
  return r;
}

// async global -> LDS byte copy (16B per lane), with portable fallback.
// ROCm clang-22 signature (from diagnostics): arg0 is
//   '__attribute__((vector_size(16))) int __device__ *'  (as(1) v4i*)
static __device__ __forceinline__ void async_copy16(const __bf16* src, __bf16* dst) {
#if __has_builtin(__builtin_amdgcn_global_load_async_to_lds_b128)
  typedef int v4i_gcc __attribute__((vector_size(16)));
  typedef __attribute__((address_space(1))) v4i_gcc* gp_t;
  typedef __attribute__((address_space(3))) v4i_gcc* lp_t;
  __builtin_amdgcn_global_load_async_to_lds_b128(
      (gp_t)(unsigned long long)src,
      (lp_t)(unsigned)(unsigned long long)dst, 0, 0);
#else
  *(v8bf*)dst = *(const v8bf*)src;
#endif
}

static __device__ __forceinline__ void async_wait0() {
#if __has_builtin(__builtin_amdgcn_global_load_async_to_lds_b128)
#if __has_builtin(__builtin_amdgcn_s_wait_asynccnt)
  __builtin_amdgcn_s_wait_asynccnt(0);
#else
  asm volatile("s_wait_asynccnt 0" ::: "memory");
#endif
#endif
}

// T5-style relative position bucket (NUM_BUCKETS=32 -> nb=16, max_exact=8)
static __device__ __forceinline__ int rel_bucket(int rel) {
  int n = -rel;
  int ret = 0;
  if (n < 0) { ret = 16; n = -n; }
  int v;
  if (n < 8) {
    v = n;
  } else {
    // 8 + log(n/8) / log(128/8) * 8 ;  8/log(16) = 2.8853901
    float lr = __logf((float)n * 0.125f);
    v = 8 + (int)(lr * 2.8853901f);
    if (v > 15) v = 15;
  }
  return ret + v;
}

// ---------------------------------------------------------------------------
// GEMM: C[M,N] = A[M,K] @ B[K,N] (+bias), bf16 WMMA, f32 accumulate.
// Compile-time N/K so epilogue addressing folds to immediates; OutT selects
// f32 or bf16 output. Double-buffered LDS overlaps staging with WMMA.
// block tile 128x128, k-step 32, 8 waves, each wave 64x32 = 4x2 WMMA tiles.
// ---------------------------------------------------------------------------
#define TM 128
#define TN 128
#define TK 32
#define LDA_S 40   // 32 + 8 pad (80B rows, 16B aligned)
#define LDB_S 40   // Bs stored transposed [n][k]

template <int N, int K, bool HAS_BIAS, typename OutT>
__global__ __launch_bounds__(256, 2)
void gemm_bf16_wmma(const float* __restrict__ A, const float* __restrict__ Bm,
                    const float* __restrict__ bias, OutT* __restrict__ C) {
  __shared__ __bf16 As[2][TM * LDA_S];
  __shared__ __bf16 Bs[2][TN * LDB_S];

  const int tid  = threadIdx.x;
  const int lane = tid & 31;
  const int wave = tid >> 5;
  const int half = lane >> 4;
  const int l16  = lane & 15;
  const int wm = wave & 1;   // 0..1 -> 64 rows
  const int wn = wave >> 1;  // 0..3 -> 32 cols

  const int bm = blockIdx.y * TM;
  const int bn = blockIdx.x * TN;

  const int arow = tid >> 3, acol = (tid & 7) * 4;   // A staging coords
  const int brow = tid >> 5, bcol = (tid & 31) * 4;  // B staging coords

  v8f acc[4][2] = {};

  // stage one k-tile (f32 -> bf16) into LDS buffer `buf`
  auto stage = [&](int k0, int buf) {
#pragma unroll
    for (int p = 0; p < 4; ++p) {
      int row = p * 32 + arow;
      float4 v = *(const float4*)&A[(size_t)(bm + row) * K + k0 + acol];
      v4bf w;
      w[0] = (__bf16)v.x; w[1] = (__bf16)v.y; w[2] = (__bf16)v.z; w[3] = (__bf16)v.w;
      *(v4bf*)&As[buf][row * LDA_S + acol] = w;
    }
#pragma unroll
    for (int p = 0; p < 4; ++p) {
      int krow = p * 8 + brow;
      float4 v = *(const float4*)&Bm[(size_t)(k0 + krow) * N + bn + bcol];
      Bs[buf][(bcol + 0) * LDB_S + krow] = (__bf16)v.x;
      Bs[buf][(bcol + 1) * LDB_S + krow] = (__bf16)v.y;
      Bs[buf][(bcol + 2) * LDB_S + krow] = (__bf16)v.z;
      Bs[buf][(bcol + 3) * LDB_S + krow] = (__bf16)v.w;
    }
  };

  constexpr int NK = K / TK;
  stage(0, 0);

  for (int kt = 0; kt < NK; ++kt) {
    const int buf = kt & 1;
    __syncthreads();
    if (kt + 1 < NK) stage((kt + 1) * TK, buf ^ 1);
    if (kt + 2 < NK) {
      __builtin_prefetch(&A[(size_t)(bm + arow) * K + (kt + 2) * TK + acol], 0, 0);
      __builtin_prefetch(&Bm[(size_t)((kt + 2) * TK + brow) * N + bn + bcol], 0, 0);
    }

    // A fragment: lane row = l16; elem e -> K = half*16 + e
    v16bf af[4];
#pragma unroll
    for (int mi = 0; mi < 4; ++mi) {
      const __bf16* pa = &As[buf][(wm * 64 + mi * 16 + l16) * LDA_S + half * 16];
      af[mi] = pack16(pa, pa + 8);
    }
    // B fragment: lane col = l16; elem e -> K = half*16 + e (contiguous Bs[n][k])
    v16bf bf[2];
#pragma unroll
    for (int ni = 0; ni < 2; ++ni) {
      const __bf16* pb = &Bs[buf][(wn * 32 + ni * 16 + l16) * LDB_S + half * 16];
      bf[ni] = pack16(pb, pb + 8);
    }
#pragma unroll
    for (int mi = 0; mi < 4; ++mi)
#pragma unroll
      for (int ni = 0; ni < 2; ++ni)
        acc[mi][ni] = __builtin_amdgcn_wmma_f32_16x16x32_bf16(
            false, af[mi], false, bf[ni], (short)0, acc[mi][ni], false, false);
  }

  // D layout: elem r -> row = r + half*8, col = l16
#pragma unroll
  for (int mi = 0; mi < 4; ++mi) {
#pragma unroll
    for (int ni = 0; ni < 2; ++ni) {
      int gcol = bn + wn * 32 + ni * 16 + l16;
      float bv = HAS_BIAS ? bias[gcol] : 0.0f;
#pragma unroll
      for (int r = 0; r < 8; ++r) {
        int grow = bm + wm * 64 + mi * 16 + r + half * 8;
        C[(size_t)grow * N + gcol] = (OutT)(acc[mi][ni][r] + bv);
      }
    }
  }
}

// ---------------------------------------------------------------------------
// Flash attention with fused bucketed rel-pos bias + elevation bias.
// qkv intermediate is bf16: K tiles stream global->LDS via the async path
// (ASYNCcnt), V tiles are transposed into LDS, Q fragments are direct 16B
// global loads (A-layout K-groups are contiguous in memory).
// One block = (b, h, 64-query tile). 4 waves, each wave = 16 query rows.
// ---------------------------------------------------------------------------
#define LDK 72   // 64 + 8 pad (144B rows)

__global__ __launch_bounds__(128, 2)
void topo_flash_attn(const __bf16* __restrict__ qkv, const float* __restrict__ coords,
                     const float* __restrict__ elev, const float* __restrict__ btab,
                     const float* __restrict__ alpha_p, float* __restrict__ ctx) {
  __shared__ float  s_bias[NBUCK * NBUCK];       // bias_table[:, h]  (4 KB)
  __shared__ int    s_qcx[64], s_qcy[64];
  __shared__ float  s_qel[64];
  __shared__ int    s_kcx[64], s_kcy[64];
  __shared__ float  s_kel[64];
  __shared__ __bf16 s_K[64 * LDK];               // [key][d]
  __shared__ __bf16 s_V[64 * LDK];               // [d][key] (transposed)
  __shared__ __bf16 s_P[4 * 16 * LDK];           // per-wave [row][key]

  const int tid  = threadIdx.x;
  const int wave = tid >> 5;
  const int lane = tid & 31;
  const int half = lane >> 4;
  const int l16  = lane & 15;
  const int qb = blockIdx.x, h = blockIdx.y, b = blockIdx.z;
  const int q0 = qb * 64;
  const float alpha = alpha_p[0];

  for (int i = tid; i < NBUCK * NBUCK; i += 128) s_bias[i] = btab[i * N_HEADS + h];
  if (tid < 64) {
    int qi = b * N_SEQ + q0 + tid;
    s_qcx[tid] = (int)(coords[qi * 2 + 0] * 128.0f);
    s_qcy[tid] = (int)(coords[qi * 2 + 1] * 128.0f);
    s_qel[tid] = elev[qi];
  }
  __syncthreads();

  // hoist loop-invariant query-side coords/elevation into registers
  int qcx_r[8], qcy_r[8];
  float qel_r[8];
#pragma unroll
  for (int r = 0; r < 8; ++r) {
    int qr = wave * 16 + r + half * 8;
    qcx_r[r] = s_qcx[qr];
    qcy_r[r] = s_qcy[qr];
    qel_r[r] = s_qel[qr];
  }

  // Q fragments (A layout): lane row = l16; elem e -> K = half*16 + e
  const __bf16* qrow =
      &qkv[((size_t)(b * N_SEQ) + q0 + wave * 16 + l16) * (3 * D_MODEL) + h * HEAD_DIM];
  v16bf qf[2];
#pragma unroll
  for (int kd = 0; kd < 2; ++kd)
    qf[kd] = pack16(qrow + kd * 32 + half * 16, qrow + kd * 32 + half * 16 + 8);

  float m_r[8], l_r[8];
  v8f oacc[4] = {};
#pragma unroll
  for (int r = 0; r < 8; ++r) { m_r[r] = -3.0e38f; l_r[r] = 0.0f; }

  for (int kb = 0; kb < N_SEQ / 64; ++kb) {
    const int j0 = kb * 64;
    // ---- K tile: async global->LDS (no transpose, raw bf16 bytes) ------
#pragma unroll
    for (int it = 0; it < 4; ++it) {
      int idx = it * 128 + tid;
      int key = idx >> 3;
      int c   = (idx & 7) * 8;
      const __bf16* src =
          &qkv[((size_t)(b * N_SEQ) + j0 + key) * (3 * D_MODEL) + D_MODEL + h * HEAD_DIM + c];
      async_copy16(src, &s_K[key * LDK + c]);
    }
    // ---- V tile: load + transpose into LDS -----------------------------
#pragma unroll
    for (int it = 0; it < 4; ++it) {
      int idx = it * 128 + tid;
      int key = idx >> 3;
      int c   = (idx & 7) * 8;
      const __bf16* src =
          &qkv[((size_t)(b * N_SEQ) + j0 + key) * (3 * D_MODEL) + 2 * D_MODEL + h * HEAD_DIM + c];
      v8bf v = *(const v8bf*)src;
#pragma unroll
      for (int e = 0; e < 8; ++e) s_V[(c + e) * LDK + key] = v[e];
    }
    if (tid < 64) {
      int ki = b * N_SEQ + j0 + tid;
      s_kcx[tid] = (int)(coords[ki * 2 + 0] * 128.0f);
      s_kcy[tid] = (int)(coords[ki * 2 + 1] * 128.0f);
      s_kel[tid] = elev[ki];
    }
    // prefetch next key block's K/V rows (bf16 row = 128B)
    if (kb + 1 < N_SEQ / 64) {
      const __bf16* nk =
          &qkv[((size_t)(b * N_SEQ) + j0 + 64 + (tid >> 1)) * (3 * D_MODEL) +
               D_MODEL + (tid & 1) * D_MODEL + h * HEAD_DIM];
      __builtin_prefetch(nk, 0, 0);
    }
    async_wait0();
    __syncthreads();

    // ---- S = Q K^T : 4 key sub-tiles x 2 k-steps -----------------------
    v8f sc[4] = {};
#pragma unroll
    for (int nt = 0; nt < 4; ++nt) {
#pragma unroll
      for (int kd = 0; kd < 2; ++kd) {
        const __bf16* pb = &s_K[(nt * 16 + l16) * LDK + kd * 32 + half * 16];
        v16bf bfrag = pack16(pb, pb + 8);
        sc[nt] = __builtin_amdgcn_wmma_f32_16x16x32_bf16(
            false, qf[kd], false, bfrag, (short)0, sc[nt], false, false);
      }
    }

    // ---- fuse scale + rel-pos bucket bias + elevation bias -------------
#pragma unroll
    for (int nt = 0; nt < 4; ++nt) {
      int kj = nt * 16 + l16;
      int kcx = s_kcx[kj], kcy = s_kcy[kj];
      float kel = s_kel[kj];
#pragma unroll
      for (int r = 0; r < 8; ++r) {
        int bx = rel_bucket(qcx_r[r] - kcx);
        int by = rel_bucket(qcy_r[r] - kcy);
        float bias = s_bias[bx * NBUCK + by];
        float ed = (kel - qel_r[r]) * 0.001f;            // (elev_j - elev_i)/1000
        float eb = fminf(0.0f, fmaxf(-10.0f, -alpha * fmaxf(ed, 0.0f)));
        sc[nt][r] = sc[nt][r] * 0.125f + bias + eb;      // scale = hd^-0.5 = 1/8
      }
    }

    // ---- online softmax ------------------------------------------------
    float mb[8];
#pragma unroll
    for (int r = 0; r < 8; ++r)
      mb[r] = fmaxf(fmaxf(sc[0][r], sc[1][r]), fmaxf(sc[2][r], sc[3][r]));
#pragma unroll
    for (int off = 1; off < 16; off <<= 1)
#pragma unroll
      for (int r = 0; r < 8; ++r)
        mb[r] = fmaxf(mb[r], __shfl_xor(mb[r], off, 32));

    float resc[8], lb[8];
#pragma unroll
    for (int r = 0; r < 8; ++r) {
      float nm = fmaxf(m_r[r], mb[r]);
      resc[r] = __expf(m_r[r] - nm);
      m_r[r] = nm;
      lb[r] = 0.0f;
    }
#pragma unroll
    for (int nt = 0; nt < 4; ++nt)
#pragma unroll
      for (int r = 0; r < 8; ++r) {
        float p = __expf(sc[nt][r] - m_r[r]);
        sc[nt][r] = p;
        lb[r] += p;
      }
#pragma unroll
    for (int off = 1; off < 16; off <<= 1)
#pragma unroll
      for (int r = 0; r < 8; ++r)
        lb[r] += __shfl_xor(lb[r], off, 32);
#pragma unroll
    for (int r = 0; r < 8; ++r) l_r[r] = l_r[r] * resc[r] + lb[r];
#pragma unroll
    for (int vt = 0; vt < 4; ++vt)
#pragma unroll
      for (int r = 0; r < 8; ++r) oacc[vt][r] *= resc[r];

    // ---- P: D layout -> per-wave LDS -> A layout -----------------------
    __bf16* pw = &s_P[wave * 16 * LDK];
#pragma unroll
    for (int nt = 0; nt < 4; ++nt)
#pragma unroll
      for (int r = 0; r < 8; ++r)
        pw[(r + half * 8) * LDK + nt * 16 + l16] = (__bf16)sc[nt][r];

    // ---- O += P V ------------------------------------------------------
#pragma unroll
    for (int kk = 0; kk < 2; ++kk) {
      const __bf16* pp = &pw[l16 * LDK + kk * 32 + half * 16];
      v16bf pf = pack16(pp, pp + 8);
#pragma unroll
      for (int vt = 0; vt < 4; ++vt) {
        const __bf16* pv = &s_V[(vt * 16 + l16) * LDK + kk * 32 + half * 16];
        v16bf vf = pack16(pv, pv + 8);
        oacc[vt] = __builtin_amdgcn_wmma_f32_16x16x32_bf16(
            false, pf, false, vf, (short)0, oacc[vt], false, false);
      }
    }
    __syncthreads();
  }

  // ---- epilogue: normalize, store ctx[b,n, h*64+d] ---------------------
#pragma unroll
  for (int vt = 0; vt < 4; ++vt) {
    int gcol = h * HEAD_DIM + vt * 16 + l16;
#pragma unroll
    for (int r = 0; r < 8; ++r) {
      int qi = q0 + wave * 16 + r + half * 8;
      ctx[((size_t)(b * N_SEQ) + qi) * D_MODEL + gcol] = oacc[vt][r] / l_r[r];
    }
  }
}

// ---------------------------------------------------------------------------
// launch
// ---------------------------------------------------------------------------
extern "C" void kernel_launch(void* const* d_in, const int* in_sizes, int n_in,
                              void* d_out, int out_size, void* d_ws, size_t ws_size,
                              hipStream_t stream) {
  const float* x      = (const float*)d_in[0];  // [B,N,D]
  const float* coords = (const float*)d_in[1];  // [B,N,2]
  const float* elevp  = (const float*)d_in[2];  // [B,N]
  const float* W_qkv  = (const float*)d_in[3];  // [D, 3D]
  const float* W_proj = (const float*)d_in[4];  // [D, D]
  const float* b_proj = (const float*)d_in[5];  // [D]
  const float* btab   = (const float*)d_in[6];  // [1024, H]
  const float* alpha  = (const float*)d_in[7];  // [1]
  float* out = (float*)d_out;

  const int M = B_SZ * N_SEQ;                                  // 4096
  __bf16* qkv = (__bf16*)d_ws;                                 // 4096 x 2304 bf16
  float*  ctx = (float*)(qkv + (size_t)M * (3 * D_MODEL));     // 4096 x 768 f32

  // 1) QKV projection: [4096,768] @ [768,2304] -> bf16
  gemm_bf16_wmma<3 * D_MODEL, D_MODEL, false, __bf16>
      <<<dim3((3 * D_MODEL) / TN, M / TM), 256, 0, stream>>>(x, W_qkv, nullptr, qkv);

  // 2) fused attention per (64-query tile, head, batch)
  topo_flash_attn<<<dim3(N_SEQ / 64, N_HEADS, B_SZ), 128, 0, stream>>>(
      qkv, coords, elevp, btab, alpha, ctx);

  // 3) output projection: [4096,768] @ [768,768] + bias -> f32
  gemm_bf16_wmma<D_MODEL, D_MODEL, true, float>
      <<<dim3(D_MODEL / TN, M / TM), 256, 0, stream>>>(ctx, W_proj, b_proj, out);
}
